// NetlistHeteroGNN_56212531970363
// MI455X (gfx1250) — matrix operand order repaired and checked by reading, not compiled, not run on hardware
//
#include <hip/hip_runtime.h>
#include <hip/hip_bf16.h>

// ---------------------------------------------------------------------------
// NetlistHeteroGNN for MI455X (gfx1250, wave32, WMMA)
//
// - Hidden state + mean aggregate stored bf16 (halves HBM traffic; the
//   102MB/51MB gather source tables fit in the 192MB L2).
// - Mean aggregation accumulated in f32 via global atomics, converted to a
//   packed bf16 mean buffer in the (bandwidth-bound) mean kernel, so the
//   WMMA GEMM inner loop is pure b128 loads + v_wmma_f32_16x16x32_bf16.
// - SAGE update fused as one K=256 GEMM: [mean | x] @ [Wl ; Wr]^T, weights
//   pre-packed into the exact WMMA B-fragment lane layout.
// - GEMM output staged through LDS so every thread does one aligned b128
//   store (instead of 8 scattered b16 stores per lane).
// ---------------------------------------------------------------------------

#define HDIM   128
#define NCELL  400000
#define NNET   200000
#define LAYERS 3

typedef __bf16          bf16x16 __attribute__((ext_vector_type(16)));
typedef unsigned short  u16x16  __attribute__((ext_vector_type(16)));
typedef float           v8f     __attribute__((ext_vector_type(8)));

union Frag { u16x16 u; bf16x16 b; };

__device__ __forceinline__ unsigned short f32_to_bf16(float f) {
  union { float f; unsigned u; } v; v.f = f;
  unsigned r = v.u + 0x7FFFu + ((v.u >> 16) & 1u);   // round-to-nearest-even
  return (unsigned short)(r >> 16);
}
__device__ __forceinline__ float bf16_to_f32(unsigned short h) {
  union { unsigned u; float f; } v; v.u = ((unsigned)h) << 16;
  return v.f;
}

// ---------------------------------------------------------------------------
// Input projection: h = relu(x @ W^T + b), x:[N,IN], W:[H,IN] -> h bf16 [N,H]
// ---------------------------------------------------------------------------
template <int IN>
__global__ void proj_kernel(const float* __restrict__ x,
                            const float* __restrict__ W,
                            const float* __restrict__ b,
                            unsigned short* __restrict__ h, int N) {
  int idx = blockIdx.x * blockDim.x + threadIdx.x;
  if (idx >= N * HDIM) return;
  int node = idx >> 7;
  int o    = idx & (HDIM - 1);
  const float* xr = x + (size_t)node * IN;
  const float* wr = W + (size_t)o * IN;
  float acc = b[o];
#pragma unroll
  for (int k = 0; k < IN; ++k) acc += xr[k] * wr[k];
  h[idx] = f32_to_bf16(acc > 0.f ? acc : 0.f);
}

// ---------------------------------------------------------------------------
// Pack combined [Wl ; Wr] (two [H,H] row-major f32) into WMMA B-fragment
// order: out[ob][kb][lane][e] bf16, with B(32x16) layout: col n = lane&15,
// k = kb*32 + (lane>>4)*8 + (e&7) + (e>=8 ? 16 : 0).
// ---------------------------------------------------------------------------
__global__ void pack_w_kernel(const float* __restrict__ Wl,
                              const float* __restrict__ Wr,
                              unsigned short* __restrict__ out) {
  int idx = blockIdx.x * blockDim.x + threadIdx.x;
  if (idx >= 8 * 8 * 32 * 16) return;
  int e    = idx & 15;
  int lane = (idx >> 4) & 31;
  int kb   = (idx >> 9) & 7;
  int ob   = idx >> 12;
  int o = ob * 16 + (lane & 15);
  int k = kb * 32 + (lane >> 4) * 8 + (e & 7) + ((e >> 3) << 4);
  float w = (k < HDIM) ? Wl[(size_t)o * HDIM + k]
                       : Wr[(size_t)o * HDIM + (k - HDIM)];
  out[idx] = f32_to_bf16(w);
}

// ---------------------------------------------------------------------------
__global__ void fill_zero_kernel(float* __restrict__ p, size_t n) {
  size_t i = (size_t)blockIdx.x * blockDim.x + threadIdx.x;
  if (i < n) p[i] = 0.f;
}

// ---------------------------------------------------------------------------
// Edge scatter-add: one wave (32 lanes) per edge, 4 features per lane.
// agg[dst] += bf16(h[src]);  deg[dst] += 1
// ---------------------------------------------------------------------------
__global__ void scatter_add_kernel(const unsigned short* __restrict__ h,
                                   const int* __restrict__ src,
                                   const int* __restrict__ dst,
                                   float* __restrict__ agg,
                                   float* __restrict__ deg, int E) {
  int t    = blockIdx.x * blockDim.x + threadIdx.x;
  int edge = t >> 5;
  if (edge >= E) return;
  int lane = t & 31;
  int s = src[edge];
  int d = dst[edge];
  const unsigned short* hp = h + (size_t)s * HDIM + lane * 4;
  float* ap = agg + (size_t)d * HDIM + lane * 4;
#pragma unroll
  for (int j = 0; j < 4; ++j) atomicAdd(ap + j, bf16_to_f32(hp[j]));
  if (lane == 0) atomicAdd(deg + d, 1.0f);
}

// ---------------------------------------------------------------------------
// mean[n][k] = bf16(agg[n][k] / max(deg[n], 1)); 4 elements per thread so
// loads are float4 and stores are packed 8B. Keeps all cvt VALU out of GEMM.
// ---------------------------------------------------------------------------
__global__ void mean_bf16_kernel(const float* __restrict__ agg,
                                 const float* __restrict__ deg,
                                 unsigned short* __restrict__ mean, int N) {
  int idx = blockIdx.x * blockDim.x + threadIdx.x;   // one per 4 elements
  if (idx >= N * (HDIM / 4)) return;
  int node = idx >> 5;
  float dg = deg[node];
  float inv = 1.0f / (dg > 1.f ? dg : 1.f);
  const float4 a = ((const float4*)agg)[idx];
  union { unsigned short us[4]; uint2 u2; } o;
  o.us[0] = f32_to_bf16(a.x * inv);
  o.us[1] = f32_to_bf16(a.y * inv);
  o.us[2] = f32_to_bf16(a.z * inv);
  o.us[3] = f32_to_bf16(a.w * inv);
  ((uint2*)mean)[idx] = o.u2;
}

// ---------------------------------------------------------------------------
// WMMA fragment loaders.
// A (16x32 bf16) layout: row M = lane&15; hi = lane>>4;
//   half e -> k = kb*32 + hi*8 + (e&7) + (e>=8 ? 16 : 0)
// B: prepacked contiguous 16 halfs per lane.
// ---------------------------------------------------------------------------
__device__ __forceinline__ bf16x16 load_a_bf16(const unsigned short* __restrict__ row,
                                               int kb, int hi) {
  int k0 = kb * 32 + hi * 8;
  Frag f;
#pragma unroll
  for (int e = 0; e < 8; ++e) f.u[e] = row[k0 + e];
#pragma unroll
  for (int e = 0; e < 8; ++e) f.u[8 + e] = row[k0 + 16 + e];
  return f.b;
}
__device__ __forceinline__ bf16x16 load_b_frag(const unsigned short* __restrict__ Wpk,
                                               int ob, int kb, int lane) {
  const unsigned short* p = Wpk + (((ob * 8 + kb) * 32 + lane) << 4);
  Frag f;
#pragma unroll
  for (int e = 0; e < 16; ++e) f.u[e] = p[e];
  return f.b;
}

// ---------------------------------------------------------------------------
// Fused SAGE GEMM: y = relu([mean | x] @ [Wl;Wr]^T + b), K = 256 via 8 WMMAs.
// Block = 256 threads = 8 waves; wave w owns output cols [16w,16w+16);
// blockIdx.x = tile of 16 nodes (N is a multiple of 16 for both node types).
// Output staged via LDS: one aligned b128 store per thread.
// ---------------------------------------------------------------------------
__global__ __launch_bounds__(256) void sage_gemm_kernel(
    const unsigned short* __restrict__ mean,  // [N,H] bf16 mean aggregate
    const unsigned short* __restrict__ xdst,  // [N,H] bf16 dst features
    const unsigned short* __restrict__ Wpk,   // packed [8][8][32][16] bf16
    const float* __restrict__ bias,           // [H]
    unsigned short* __restrict__ y,           // [N,H] bf16 out
    int N) {
  __shared__ __align__(16) unsigned short tile_out[16 * HDIM];  // 4KB

  int wave = threadIdx.x >> 5;      // out-block 0..7
  int lane = threadIdx.x & 31;
  int hi   = lane >> 4;
  int tile = blockIdx.x;
  int row  = tile * 16 + (lane & 15);
  int n    = wave * 16 + (lane & 15);

  float bv = bias[n];
  v8f c;
#pragma unroll
  for (int r = 0; r < 8; ++r) c[r] = bv;

  const unsigned short* mrow = mean + (size_t)row * HDIM;
  const unsigned short* xrow = xdst + (size_t)row * HDIM;

#pragma unroll
  for (int kb = 0; kb < 4; ++kb) {      // K 0..127: mean aggregate x Wl
    bf16x16 a = load_a_bf16(mrow, kb, hi);
    bf16x16 b = load_b_frag(Wpk, wave, kb, lane);
    c = __builtin_amdgcn_wmma_f32_16x16x32_bf16(false, a, false, b,
                                                (short)0, c, false, false);
  }
#pragma unroll
  for (int kb = 0; kb < 4; ++kb) {      // K 128..255: dst features x Wr
    bf16x16 a = load_a_bf16(xrow, kb, hi);
    bf16x16 b = load_b_frag(Wpk, wave, kb + 4, lane);
    c = __builtin_amdgcn_wmma_f32_16x16x32_bf16(false, a, false, b,
                                                (short)0, c, false, false);
  }

  // D layout: VGPR r, lanes 0-15: M=r; lanes 16-31: M=8+r; N=lane&15.
  // Stage relu'd bf16 tile in LDS for coalesced global stores.
#pragma unroll
  for (int r = 0; r < 8; ++r) {
    float v = c[r];
    tile_out[(hi * 8 + r) * HDIM + n] = f32_to_bf16(v > 0.f ? v : 0.f);
  }
  __syncthreads();

  // 16 rows x 128 cols x 2B = 4096B = 256 threads x one uint4 each.
  uint4 val = ((const uint4*)tile_out)[threadIdx.x];
  ((uint4*)(y + (size_t)tile * 16 * HDIM))[threadIdx.x] = val;
}

// ---------------------------------------------------------------------------
// Output head: out = hc @ Wh^T + bh, OUT = 2
// ---------------------------------------------------------------------------
__global__ void head_kernel(const unsigned short* __restrict__ hc,
                            const float* __restrict__ Wh,
                            const float* __restrict__ bh,
                            float* __restrict__ out, int N) {
  int node = blockIdx.x * blockDim.x + threadIdx.x;
  if (node >= N) return;
  const unsigned short* hr = hc + (size_t)node * HDIM;
  float a0 = bh[0], a1 = bh[1];
#pragma unroll 8
  for (int k = 0; k < HDIM; ++k) {
    float v = bf16_to_f32(hr[k]);
    a0 += v * Wh[k];
    a1 += v * Wh[HDIM + k];
  }
  out[node * 2 + 0] = a0;
  out[node * 2 + 1] = a1;
}

// ---------------------------------------------------------------------------
extern "C" void kernel_launch(void* const* d_in, const int* in_sizes, int n_in,
                              void* d_out, int out_size, void* d_ws, size_t ws_size,
                              hipStream_t stream) {
  const float* x_cell = (const float*)d_in[0];
  const float* x_net  = (const float*)d_in[1];
  const int*   e_drv  = (const int*)d_in[2];   // [2, E]: row0 src(cell), row1 dst(net)
  const int*   e_fan  = (const int*)d_in[3];   // [2, E]: row0 src(net),  row1 dst(cell)
  const float* Wc     = (const float*)d_in[4];
  const float* bc     = (const float*)d_in[5];
  const float* Wn     = (const float*)d_in[6];
  const float* bn     = (const float*)d_in[7];
  const float* Wl_dn  = (const float*)d_in[8];
  const float* bl_dn  = (const float*)d_in[9];
  const float* Wr_dn  = (const float*)d_in[10];
  const float* Wl_nc  = (const float*)d_in[11];
  const float* bl_nc  = (const float*)d_in[12];
  const float* Wr_nc  = (const float*)d_in[13];
  const float* Wh     = (const float*)d_in[14];
  const float* bh     = (const float*)d_in[15];
  float* out = (float*)d_out;

  const int E_drv = in_sizes[2] / 2;
  const int E_fan = in_sizes[3] / 2;

  // ---- workspace carve-up ----
  char* ws = (char*)d_ws;
  size_t off = 0;
  auto carve = [&](size_t bytes) -> void* {
    void* p = ws + off;
    off = (off + bytes + 255) & ~(size_t)255;
    return p;
  };
  unsigned short* hcbuf[2];
  unsigned short* hnbuf[2];
  hcbuf[0] = (unsigned short*)carve((size_t)NCELL * HDIM * 2);
  hcbuf[1] = (unsigned short*)carve((size_t)NCELL * HDIM * 2);
  hnbuf[0] = (unsigned short*)carve((size_t)NNET * HDIM * 2);
  hnbuf[1] = (unsigned short*)carve((size_t)NNET * HDIM * 2);
  float* agg_c = (float*)carve((size_t)NCELL * HDIM * 4);
  float* agg_n = (float*)carve((size_t)NNET * HDIM * 4);
  unsigned short* mean_c = (unsigned short*)carve((size_t)NCELL * HDIM * 2);
  unsigned short* mean_n = (unsigned short*)carve((size_t)NNET * HDIM * 2);
  float* deg_c = (float*)carve((size_t)NCELL * 4);
  float* deg_n = (float*)carve((size_t)NNET * 4);
  const size_t PKSZ = 8 * 8 * 32 * 16;  // elements per packed weight pair
  unsigned short* pk_dn[LAYERS];
  unsigned short* pk_nc[LAYERS];
  for (int i = 0; i < LAYERS; ++i) pk_dn[i] = (unsigned short*)carve(PKSZ * 2);
  for (int i = 0; i < LAYERS; ++i) pk_nc[i] = (unsigned short*)carve(PKSZ * 2);

  // ---- pack combined [Wl;Wr] weights into WMMA B-fragment order ----
  {
    dim3 g((unsigned)((PKSZ + 255) / 256)), b(256);
    for (int i = 0; i < LAYERS; ++i) {
      pack_w_kernel<<<g, b, 0, stream>>>(Wl_dn + (size_t)i * HDIM * HDIM,
                                         Wr_dn + (size_t)i * HDIM * HDIM, pk_dn[i]);
      pack_w_kernel<<<g, b, 0, stream>>>(Wl_nc + (size_t)i * HDIM * HDIM,
                                         Wr_nc + (size_t)i * HDIM * HDIM, pk_nc[i]);
    }
  }

  // ---- input projections ----
  proj_kernel<16><<<(NCELL * HDIM + 255) / 256, 256, 0, stream>>>(
      x_cell, Wc, bc, hcbuf[0], NCELL);
  proj_kernel<8><<<(NNET * HDIM + 255) / 256, 256, 0, stream>>>(
      x_net, Wn, bn, hnbuf[0], NNET);

  unsigned short* hc_cur = hcbuf[0];
  unsigned short* hc_nxt = hcbuf[1];
  unsigned short* hn_cur = hnbuf[0];
  unsigned short* hn_nxt = hnbuf[1];

  for (int i = 0; i < LAYERS; ++i) {
    // zero aggregation buffers + degrees
    fill_zero_kernel<<<(unsigned)(((size_t)NNET * HDIM + 255) / 256), 256, 0, stream>>>(
        agg_n, (size_t)NNET * HDIM);
    fill_zero_kernel<<<(NNET + 255) / 256, 256, 0, stream>>>(deg_n, (size_t)NNET);
    fill_zero_kernel<<<(unsigned)(((size_t)NCELL * HDIM + 255) / 256), 256, 0, stream>>>(
        agg_c, (size_t)NCELL * HDIM);
    fill_zero_kernel<<<(NCELL + 255) / 256, 256, 0, stream>>>(deg_c, (size_t)NCELL);

    // scatter: drives (cell -> net), fans (net -> cell), both from OLD states
    scatter_add_kernel<<<(unsigned)(((size_t)E_drv * 32 + 255) / 256), 256, 0, stream>>>(
        hc_cur, e_drv, e_drv + E_drv, agg_n, deg_n, E_drv);
    scatter_add_kernel<<<(unsigned)(((size_t)E_fan * 32 + 255) / 256), 256, 0, stream>>>(
        hn_cur, e_fan, e_fan + E_fan, agg_c, deg_c, E_fan);

    // mean (f32 accumulate -> packed bf16 mean buffer)
    mean_bf16_kernel<<<(NNET * (HDIM / 4) + 255) / 256, 256, 0, stream>>>(
        agg_n, deg_n, mean_n, NNET);
    mean_bf16_kernel<<<(NCELL * (HDIM / 4) + 255) / 256, 256, 0, stream>>>(
        agg_c, deg_c, mean_c, NCELL);

    // fused SAGE GEMMs (WMMA)
    sage_gemm_kernel<<<NNET / 16, 256, 0, stream>>>(
        mean_n, hn_cur, pk_dn[i], bl_dn + (size_t)i * HDIM, hn_nxt, NNET);
    sage_gemm_kernel<<<NCELL / 16, 256, 0, stream>>>(
        mean_c, hc_cur, pk_nc[i], bl_nc + (size_t)i * HDIM, hc_nxt, NCELL);

    // swap
    unsigned short* t;
    t = hc_cur; hc_cur = hc_nxt; hc_nxt = t;
    t = hn_cur; hn_cur = hn_nxt; hn_nxt = t;
  }

  // ---- output head ----
  head_kernel<<<(NCELL + 255) / 256, 256, 0, stream>>>(hc_cur, Wh, bh, out, NCELL);
}
